// XNORNetBlock_78108275245424
// MI455X (gfx1250) — 8-bit matrix operand order repaired and checked by reading, compiled, not run on hardware
//
#include <hip/hip_runtime.h>
#include <stdint.h>


typedef __attribute__((ext_vector_type(8))) int          v8i;
typedef __attribute__((ext_vector_type(4))) unsigned int u32x4;
typedef __attribute__((ext_vector_type(4))) float        f32x4;
typedef unsigned long long ull;

#define NB   16
#define CIN  256
#define COUT 128
#define HH   128
#define WW   128
#define XH   130
#define XW   130

#define ALPHA_OFF 0u
#define WFRAG_OFF 512u
#define XS_OFF    (1u << 20)

__global__ void alpha_kernel(const float* __restrict__ w, float* __restrict__ alpha) {
    __shared__ float red[256];
    const int co = blockIdx.x;
    const int t  = threadIdx.x;
    float s = 0.f;
    for (int i = t; i < CIN * 9; i += 256) s += fabsf(w[co * CIN * 9 + i]);
    red[t] = s;
    __syncthreads();
    for (int k = 128; k > 0; k >>= 1) {
        if (t < k) red[t] += red[t + k];
        __syncthreads();
    }
    if (t == 0) alpha[co] = red[0] * (1.0f / (float)(CIN * 9));
}

__global__ void pack_w_kernel(const float* __restrict__ w, uint32_t* __restrict__ wfrag) {
    const uint32_t flat = blockIdx.x * 256u + threadIdx.x;
    const uint32_t v    = flat & 7u;
    const uint32_t lane = (flat >> 3) & 31u;
    const uint32_t kc   = (flat >> 8) & 3u;
    const uint32_t ct   = (flat >> 10) & 7u;
    const uint32_t kk   = flat >> 13;
    const uint32_t kh   = kk / 3u, kw = kk % 3u;
    const uint32_t kb   = ((v >> 2) << 5) | (((lane >> 4) & 1u) << 4) | ((v & 3u) << 2);
    const uint32_t co   = ct * 16u + (lane & 15u);
    uint32_t d = 0;
#pragma unroll
    for (int b = 0; b < 4; ++b) {
        const uint32_t ci = kc * 64u + kb + (uint32_t)b;
        const float    x  = w[((co * CIN + ci) * 3u + kh) * 3u + kw];
        const int      s  = (x > 0.f) - (x < 0.f);
        d |= ((uint32_t)(uint8_t)(int8_t)s) << (8 * b);
    }
    wfrag[flat] = d;
}

__global__ void zero_pad_kernel(int8_t* __restrict__ xs) {
    const int bx = blockIdx.x;
    const int n  = bx / XH;
    const int hp = bx % XH;
    const int t  = threadIdx.x;
    u32x4* row = (u32x4*)(xs + ((size_t)n * XH + hp) * XW * CIN);
    const u32x4 z = {0u, 0u, 0u, 0u};
    if (hp == 0 || hp == XH - 1) {
        for (int i = t; i < XW * CIN / 16; i += 256) row[i] = z;
    } else if (t < 32) {
        if (t < 16) row[t] = z;
        else        row[(XW - 1) * 16 + (t - 16)] = z;
    }
}

__global__ void pack_x_kernel(const float* __restrict__ x, int8_t* __restrict__ xs) {
    __shared__ int8_t tile[64 * 64];
    const int t  = threadIdx.x;
    const int n  = blockIdx.x >> 7;
    const int h  = blockIdx.x & 127;
    const int wt = blockIdx.y >> 2;
    const int ct = blockIdx.y & 3;
    const int w0 = wt * 64, ci0 = ct * 64;

    const int wl = t & 63;
    const int cs = t >> 6;
#pragma unroll
    for (int j = 0; j < 16; ++j) {
        const int   cil = j * 4 + cs;
        const float v   = __builtin_nontemporal_load(
            &x[(((size_t)n * CIN + ci0 + cil) * HH + h) * WW + w0 + wl]);
        tile[wl * 64 + cil] = (int8_t)((v > 0.f) - (v < 0.f));
    }
    __syncthreads();
    const int wl2 = t >> 2;
    const int q   = t & 3;
    const u32x4 val = *(const u32x4*)&tile[wl2 * 64 + q * 16];
    *(u32x4*)&xs[(((size_t)n * XH + (h + 1)) * XW + (w0 + wl2 + 1)) * CIN + ci0 + q * 16] = val;
}

struct AF { union { v8i v; ull q[4]; }; };
struct BF { union { v8i v; u32x4 x[2]; }; };

__launch_bounds__(256)
__global__ void conv_kernel(const int8_t* __restrict__ xs,
                            const uint32_t* __restrict__ wfrag,
                            const float* __restrict__ alpha,
                            float* __restrict__ out) {
    extern __shared__ char smem[];
    u32x4* __restrict__ smem4 = (u32x4*)smem;

    const int t    = threadIdx.x;
    const int lane = t & 31;
    const int wv   = t >> 5;
    const int n    = blockIdx.x >> 7;
    const int h    = blockIdx.x & 127;
    const int cb   = blockIdx.y;

    {
        const ull gbase = (ull)(uintptr_t)wfrag;
#pragma unroll
        for (int j = 0; j < 36; ++j) {
            const int idx    = j * 256 + t;
            const int fl     = idx >> 6;
            const int within = idx & 63;
            const int kk     = fl >> 4;
            const int ctl    = (fl >> 2) & 3;
            const int kc     = fl & 3;
            const int fg     = ((kk * 8 + cb * 4 + ctl) * 4 + kc);
            const unsigned voff = (unsigned)((fg * 64 + within) * 16);
            const unsigned ldsa = (unsigned)(idx * 16);
            asm volatile("global_load_async_to_lds_b128 %0, %1, %2"
                         :: "v"(ldsa), "v"(voff), "s"(gbase) : "memory");
        }
        asm volatile("s_wait_asynccnt 0x0" ::: "memory");
    }
    __syncthreads();

    v8i acc[4] = {};

    const int m    = lane & 15;
    const int koff = (lane & 16) ? 16 : 0;

    const int8_t* __restrict__ abase =
        xs + (((size_t)n * XH + h) * XW + (size_t)(wv * 16 + m)) * CIN + koff;

    auto load_step = [&](int s, AF& a, BF* b) {
        const int kk = s >> 2, kc = s & 3;
        const int kh = kk / 3, kw = kk % 3;
        const int off = (kh * XW + kw) * CIN + kc * 64;
        a.q[0] = *(const ull*)(abase + off);
        a.q[1] = *(const ull*)(abase + off + 8);
        a.q[2] = *(const ull*)(abase + off + 32);
        a.q[3] = *(const ull*)(abase + off + 40);
#pragma unroll
        for (int ct = 0; ct < 4; ++ct) {
            const u32x4* bp = smem4 + (((kk * 4 + ct) * 4 + kc) * 32 + lane) * 2;
            b[ct].x[0] = bp[0];
            b[ct].x[1] = bp[1];
        }
    };

    AF a[2];
    BF b[2][4];
    load_step(0, a[0], b[0]);
#pragma unroll
    for (int s = 0; s < 36; ++s) {
        const int cur = s & 1;
        if (s + 1 < 36) load_step(s + 1, a[cur ^ 1], b[cur ^ 1]);
#pragma unroll
        for (int ct = 0; ct < 4; ++ct) {
            acc[ct] = __builtin_amdgcn_wmma_i32_16x16x64_iu8(
                 true, a[cur].v,  true, b[cur][ct].v,
                acc[ct],  false,  false);
        }
    }

    __syncthreads();
    float* __restrict__ smemF = (float*)smem;
#pragma unroll
    for (int ct = 0; ct < 4; ++ct) {
        const int col   = ct * 16 + m;
        const int wbase = wv * 16 + ((lane >> 4) << 3);
#pragma unroll
        for (int r = 0; r < 8; ++r)
            smemF[col * 128 + wbase + r] = (float)acc[ct][r];
    }
    __syncthreads();

    const int    col   = t >> 2;
    const int    q     = t & 3;
    const int    cog   = cb * 64 + col;
    const size_t obase = (((size_t)n * COUT + cog) * HH + h) * WW;
#pragma unroll
    for (int j = 0; j < 8; ++j) {
        const int w  = q * 32 + j * 4;
        f32x4 vv = *(const f32x4*)&smemF[col * 128 + w];
        const f32x4 al = *(const f32x4*)&alpha[w];
        vv *= al;
        __builtin_nontemporal_store(vv, (f32x4*)&out[obase + w]);
    }
}

extern "C" void kernel_launch(void* const* d_in, const int* in_sizes, int n_in,
                              void* d_out, int out_size, void* d_ws, size_t ws_size,
                              hipStream_t stream) {
    const float* x   = (const float*)d_in[0];
    const float* w   = (const float*)d_in[1];
    float*       out = (float*)d_out;

    char*     ws    = (char*)d_ws;
    float*    alpha = (float*)(ws + ALPHA_OFF);
    uint32_t* wfrag = (uint32_t*)(ws + WFRAG_OFF);
    int8_t*   xs    = (int8_t*)(ws + XS_OFF);

    alpha_kernel<<<COUT, 256, 0, stream>>>(w, alpha);
    pack_w_kernel<<<288, 256, 0, stream>>>(w, wfrag);
    zero_pad_kernel<<<NB * XH, 256, 0, stream>>>(xs);
    pack_x_kernel<<<dim3(NB * HH, 8), 256, 0, stream>>>(x, xs);
    conv_kernel<<<dim3(NB * HH, 2), 256, 147456, stream>>>(xs, wfrag, alpha, out);
}
